// SelfAttention_21921513079472
// MI455X (gfx1250) — compile-verified, matrix operand order checked by reading, not run on hardware
//
#include <hip/hip_runtime.h>
#include <hip/hip_bf16.h>
#include <math.h>

// Problem constants (from reference): B=8, S=2048, D=1024, H=1024
#define BB 8
#define SS 2048
#define DD 1024
#define HH 1024
#define BSR (BB * SS)   // 16384 total rows

typedef __attribute__((ext_vector_type(16))) __bf16       v16bf;
typedef __attribute__((ext_vector_type(8)))  float        v8f;
typedef __attribute__((ext_vector_type(4)))  unsigned int v4u;

// ---------------------------------------------------------------------------
// Fragment layout (ISA 16-bit A/B 16x32):
//   lane L: row/col = L & 15, half = L >> 4
//   elements 0..7  <- K = k0 + half*8 + 0..7
//   elements 8..15 <- K = k0 + 16 + half*8 + 0..7
// `base` points at (row, k0 + half*8); both 16-byte chunks are aligned.
// ---------------------------------------------------------------------------

union FragU {
    v4u   u[2];
    v16bf f;
};

union PackU {
    v4u    u;
    __bf16 h[8];
};

// 8+8 contiguous bf16 -> two 16B loads (global_load_b128)
__device__ __forceinline__ v16bf frag_ld_bf16(const __bf16* __restrict__ p) {
    FragU fr;
    fr.u[0] = *(const v4u*)p;
    fr.u[1] = *(const v4u*)(p + 16);
    return fr.f;
}

// 8+8 contiguous f32, converted to bf16 (v_cvt_pk_bf16_f32)
__device__ __forceinline__ v16bf frag_cvt_f32(const float* __restrict__ p) {
    v16bf f;
#pragma unroll
    for (int j = 0; j < 8; ++j) {
        f[j]     = (__bf16)p[j];
        f[8 + j] = (__bf16)p[16 + j];
    }
    return f;
}

#define WMMA_BF16(a, b, c) \
    __builtin_amdgcn_wmma_f32_16x16x32_bf16(false, (a), false, (b), (short)0, (c), false, false)

// ===========================================================================
// Kernel 0: one-shot weight transpose + convert.
// W{q,k,v} f32 [DD][HH]  ->  WT bf16 [3][HH][DD]  (6 MB total, L2-resident).
// 64x64 tiles via LDS (stride 65 to dodge bank conflicts).
// ===========================================================================
__global__ __launch_bounds__(256) void wt_kernel(
    const float* __restrict__ Wq, const float* __restrict__ Wk,
    const float* __restrict__ Wv, __bf16* __restrict__ WT)
{
    __shared__ __bf16 Ts[64 * 65];

    const int t   = threadIdx.x;
    const int TPM = (DD / 64) * (HH / 64);       // 256 tiles per matrix
    const int mat = blockIdx.x / TPM;
    const int rem = blockIdx.x % TPM;
    const int kb  = (rem / (HH / 64)) * 64;
    const int nb  = (rem % (HH / 64)) * 64;

    const float* W = (mat == 0) ? Wq : (mat == 1) ? Wk : Wv;
    __bf16* dst = WT + (size_t)mat * HH * DD;

#pragma unroll
    for (int i = 0; i < 16; ++i) {               // read [k][n] coalesced in n
        const int idx = t + i * 256;             // 0..4095
        const int kk  = idx >> 6;
        const int nn  = idx & 63;
        Ts[nn * 65 + kk] = (__bf16)W[(size_t)(kb + kk) * HH + nb + nn];
    }
    __syncthreads();
#pragma unroll
    for (int i = 0; i < 16; ++i) {               // write [n][k] coalesced in k
        const int idx = t + i * 256;
        const int nn  = idx >> 6;
        const int kk  = idx & 63;
        dst[(size_t)(nb + nn) * DD + kb + kk] = Ts[nn * 65 + kk];
    }
}

// ===========================================================================
// Kernel 1: q,k,v = x @ W{q,k,v} + b  (f32 in -> bf16 out in workspace)
// One wave per 64x64 tile, 4x4 WMMA blocking, all operands contiguous:
//   A = x rows (f32 -> bf16 cvt), B = WT rows (bf16, b128 loads).
// q stored * 1/sqrt(H) (folds the score scale); v stored TRANSPOSED [H][S]
// so kernel 4's B operand is contiguous too.
// ===========================================================================
__global__ __launch_bounds__(256) void qkv_proj_kernel(
    const float* __restrict__ x,
    const __bf16* __restrict__ WT,
    const float* __restrict__ bq, const float* __restrict__ bk,
    const float* __restrict__ bv,
    __bf16* __restrict__ ws)   // q[BSR][HH], k[BSR][HH], vT[BB][HH][SS]
{
    const int wave = blockIdx.x * 8 + (threadIdx.x >> 5);
    const int lane = threadIdx.x & 31;
    const int half = lane >> 4;
    const int ln   = lane & 15;

    const int perMat = (BSR / 64) * (HH / 64);   // 4096 tiles per matrix
    const int mat = wave / perMat;
    const int rem = wave % perMat;
    const int row0 = (rem / (HH / 64)) * 64;     // global row (b*SS + s)
    const int col0 = (rem % (HH / 64)) * 64;

    const __bf16* Wt   = WT + (size_t)mat * HH * DD;
    const float*  bias = (mat == 0) ? bq : (mat == 1) ? bk : bv;

    v8f acc[4][4] = {};

#pragma unroll 2
    for (int k0 = 0; k0 < DD; k0 += 32) {
        v16bf afr[4], bfr[4];
#pragma unroll
        for (int mi = 0; mi < 4; ++mi)
            afr[mi] = frag_cvt_f32(x + (size_t)(row0 + mi * 16 + ln) * DD + k0 + half * 8);
#pragma unroll
        for (int ni = 0; ni < 4; ++ni)
            bfr[ni] = frag_ld_bf16(Wt + (size_t)(col0 + ni * 16 + ln) * DD + k0 + half * 8);

#pragma unroll
        for (int mi = 0; mi < 4; ++mi)
#pragma unroll
            for (int ni = 0; ni < 4; ++ni)
                acc[mi][ni] = WMMA_BF16(afr[mi], bfr[ni], acc[mi][ni]);
    }

    float bz[4];
#pragma unroll
    for (int ni = 0; ni < 4; ++ni) bz[ni] = bias[col0 + ni * 16 + ln];

    if (mat == 2) {
        // v: store transposed, vT[b][h][s]; 8 consecutive s per lane -> b128 store
        const int b    = row0 >> 11;             // row0 / SS
        const int srow = row0 & (SS - 1);
        __bf16* vT = ws + 2 * (size_t)BSR * HH + (size_t)b * HH * SS;
#pragma unroll
        for (int mi = 0; mi < 4; ++mi)
#pragma unroll
            for (int ni = 0; ni < 4; ++ni) {
                const int n = col0 + ni * 16 + ln;
                const int s = srow + mi * 16 + half * 8;
                PackU st;
#pragma unroll
                for (int r = 0; r < 8; ++r) st.h[r] = (__bf16)(acc[mi][ni][r] + bz[ni]);
                *(v4u*)&vT[(size_t)n * SS + s] = st.u;
            }
    } else {
        const float scl = (mat == 0) ? 0.03125f : 1.0f;   // fold 1/sqrt(1024) into q
        __bf16* dst = ws + (size_t)mat * BSR * HH;
#pragma unroll
        for (int mi = 0; mi < 4; ++mi)
#pragma unroll
            for (int ni = 0; ni < 4; ++ni)
#pragma unroll
                for (int r = 0; r < 8; ++r) {
                    const int m = row0 + mi * 16 + half * 8 + r;
                    const int n = col0 + ni * 16 + ln;
                    dst[(size_t)m * HH + n] = (__bf16)((acc[mi][ni][r] + bz[ni]) * scl);
                }
    }
}

// ===========================================================================
// Kernel 2: scores = (q*scale) @ k^T  (bf16 in -> f32 scores)
// Both operands contiguous in K(=h); one wave per 64x64 tile, 4x4 blocking.
// ===========================================================================
__global__ __launch_bounds__(256) void scores_kernel(
    const __bf16* __restrict__ ws,
    float* __restrict__ att)     // [BB][SS][SS]
{
    const int wave = blockIdx.x * 8 + (threadIdx.x >> 5);
    const int lane = threadIdx.x & 31;
    const int half = lane >> 4;
    const int ln   = lane & 15;

    const int perB = (SS / 64) * (SS / 64);      // 1024 tiles per batch
    const int b   = wave / perB;
    const int rem = wave % perB;
    const int m0  = (rem / (SS / 64)) * 64;
    const int n0  = (rem % (SS / 64)) * 64;

    const __bf16* qb = ws + (size_t)b * SS * HH;
    const __bf16* kb = ws + (size_t)BSR * HH + (size_t)b * SS * HH;

    v8f acc[4][4] = {};

#pragma unroll 2
    for (int k0 = 0; k0 < HH; k0 += 32) {
        v16bf afr[4], bfr[4];
#pragma unroll
        for (int mi = 0; mi < 4; ++mi)
            afr[mi] = frag_ld_bf16(qb + (size_t)(m0 + mi * 16 + ln) * HH + k0 + half * 8);
#pragma unroll
        for (int ni = 0; ni < 4; ++ni)   // col n of k^T == row n of k (contiguous)
            bfr[ni] = frag_ld_bf16(kb + (size_t)(n0 + ni * 16 + ln) * HH + k0 + half * 8);

#pragma unroll
        for (int mi = 0; mi < 4; ++mi)
#pragma unroll
            for (int ni = 0; ni < 4; ++ni)
                acc[mi][ni] = WMMA_BF16(afr[mi], bfr[ni], acc[mi][ni]);
    }

#pragma unroll
    for (int mi = 0; mi < 4; ++mi)
#pragma unroll
        for (int ni = 0; ni < 4; ++ni)
#pragma unroll
            for (int r = 0; r < 8; ++r) {
                const int m = m0 + mi * 16 + half * 8 + r;
                const int n = n0 + ni * 16 + ln;
                att[((size_t)b * SS + m) * SS + n] = acc[mi][ni][r];
            }
}

// ===========================================================================
// Kernel 3: row softmax, in place (one 256-thread block per row of 2048)
// ===========================================================================
__global__ __launch_bounds__(256) void softmax_kernel(float* __restrict__ att)
{
    __shared__ float red[256];
    const int t = threadIdx.x;
    float* row = att + (size_t)blockIdx.x * SS;

    float v[8];
    float m = -INFINITY;
#pragma unroll
    for (int i = 0; i < 8; ++i) {
        v[i] = row[t + i * 256];
        m = fmaxf(m, v[i]);
    }
    red[t] = m;
    __syncthreads();
    for (int s = 128; s > 0; s >>= 1) {
        if (t < s) red[t] = fmaxf(red[t], red[t + s]);
        __syncthreads();
    }
    m = red[0];
    __syncthreads();

    float sum = 0.f;
#pragma unroll
    for (int i = 0; i < 8; ++i) {
        v[i] = __expf(v[i] - m);
        sum += v[i];
    }
    red[t] = sum;
    __syncthreads();
    for (int s = 128; s > 0; s >>= 1) {
        if (t < s) red[t] += red[t + s];
        __syncthreads();
    }
    const float inv = 1.0f / red[0];

#pragma unroll
    for (int i = 0; i < 8; ++i)
        row[t + i * 256] = v[i] * inv;
}

// ===========================================================================
// Kernel 4: out = P @ V using vT [H][S]: both operands contiguous in K(=s).
// A = P f32 -> bf16 cvt; B = vT rows, bf16 b128 loads. 4x4 blocking, no LDS.
// ===========================================================================
__global__ __launch_bounds__(256) void av_kernel(
    const float* __restrict__ att,
    const __bf16* __restrict__ ws,
    float* __restrict__ out)    // [BB][SS][HH]
{
    const int wave = blockIdx.x * 8 + (threadIdx.x >> 5);
    const int lane = threadIdx.x & 31;
    const int half = lane >> 4;
    const int ln   = lane & 15;

    const int perB = (SS / 64) * (HH / 64);      // 512 tiles per batch
    const int b   = wave / perB;
    const int rem = wave % perB;
    const int m0  = (rem / (HH / 64)) * 64;
    const int n0  = (rem % (HH / 64)) * 64;

    const float*  Pb = att + (size_t)b * SS * SS;
    const __bf16* vT = ws + 2 * (size_t)BSR * HH + (size_t)b * HH * SS;

    v8f acc[4][4] = {};

#pragma unroll 2
    for (int k0 = 0; k0 < SS; k0 += 32) {
        v16bf afr[4], bfr[4];
#pragma unroll
        for (int mi = 0; mi < 4; ++mi)
            afr[mi] = frag_cvt_f32(Pb + (size_t)(m0 + mi * 16 + ln) * SS + k0 + half * 8);
#pragma unroll
        for (int ni = 0; ni < 4; ++ni)   // col n of V == row n of vT (contiguous)
            bfr[ni] = frag_ld_bf16(vT + (size_t)(n0 + ni * 16 + ln) * SS + k0 + half * 8);

#pragma unroll
        for (int mi = 0; mi < 4; ++mi)
#pragma unroll
            for (int ni = 0; ni < 4; ++ni)
                acc[mi][ni] = WMMA_BF16(afr[mi], bfr[ni], acc[mi][ni]);
    }

#pragma unroll
    for (int mi = 0; mi < 4; ++mi)
#pragma unroll
        for (int ni = 0; ni < 4; ++ni)
#pragma unroll
            for (int r = 0; r < 8; ++r) {
                const int m = m0 + mi * 16 + half * 8 + r;
                const int n = n0 + ni * 16 + ln;
                out[((size_t)b * SS + m) * HH + n] = acc[mi][ni][r];
            }
}

// ---------------- host entry ----------------
extern "C" void kernel_launch(void* const* d_in, const int* in_sizes, int n_in,
                              void* d_out, int out_size, void* d_ws, size_t ws_size,
                              hipStream_t stream) {
    (void)in_sizes; (void)n_in; (void)out_size; (void)ws_size;

    const float* x  = (const float*)d_in[0];
    const float* Wq = (const float*)d_in[1];
    const float* bq = (const float*)d_in[2];
    const float* Wk = (const float*)d_in[3];
    const float* bk = (const float*)d_in[4];
    const float* Wv = (const float*)d_in[5];
    const float* bv = (const float*)d_in[6];

    float* out = (float*)d_out;                       // [BB,SS,HH] weighted sum
    float* att = out + (size_t)BSR * HH;              // [BB,SS,SS] attention weights

    // workspace: q | k | vT  (3 * BSR*HH bf16 ≈ 100 MB)  then WT (3*HH*DD bf16 = 6 MB)
    __bf16* ws = (__bf16*)d_ws;
    __bf16* WT = ws + 3 * (size_t)BSR * HH;

    wt_kernel<<<3 * (DD / 64) * (HH / 64), 256, 0, stream>>>(Wq, Wk, Wv, WT);

    qkv_proj_kernel<<<(3 * (BSR / 64) * (HH / 64)) / 8, 256, 0, stream>>>(
        x, WT, bq, bk, bv, ws);

    scores_kernel<<<(BB * (SS / 64) * (SS / 64)) / 8, 256, 0, stream>>>(ws, att);

    softmax_kernel<<<BSR, 256, 0, stream>>>(att);

    av_kernel<<<(BB * (SS / 64) * (HH / 64)) / 8, 256, 0, stream>>>(att, ws, out);
}
